// WaveTFTForecaster_37065567764966
// MI455X (gfx1250) — matrix operand order repaired and checked by reading, not compiled, not run on hardware
//
#include <hip/hip_runtime.h>
#include <hip/hip_bf16.h>
#include <math.h>

// ---------------------------------------------------------------------------
// Model dims (compile-time constants from the reference)
// ---------------------------------------------------------------------------
#define BB    128
#define LL    128
#define DIN   6
#define DD    256
#define HH    8
#define NLAY  6
#define NEXP  8
#define DFF   1024
#define NOUTP 5
#define NMOE  3
#define TT    (BB * LL)     // 16384 token rows
#define DK    32            // DD / HH

typedef _Float16 half8  __attribute__((ext_vector_type(8)));
typedef _Float16 half16 __attribute__((ext_vector_type(16)));
typedef float    floatx8 __attribute__((ext_vector_type(8)));
typedef int      intx4   __attribute__((ext_vector_type(4)));

// ---------------------------------------------------------------------------
// CDNA5 async global->LDS staging (ASYNCcnt path), with sync fallback.
// Builtin signature (from hipcc diagnostic): first param is a global
// (address_space(1)) int4*, second the LDS (address_space(3)) int4*.
// ---------------------------------------------------------------------------
#if defined(__gfx1250__) && __has_builtin(__builtin_amdgcn_global_load_async_to_lds_b128)
#define ASYNC_STAGE 1
#else
#define ASYNC_STAGE 0
#endif

static __device__ inline void stage16(_Float16* dst, const _Float16* src) {
#if ASYNC_STAGE
    __builtin_amdgcn_global_load_async_to_lds_b128(
        (__attribute__((address_space(1))) intx4*)(uintptr_t)src,
        (__attribute__((address_space(3))) intx4*)(uint32_t)(uintptr_t)dst,
        0, 0);
#else
    *(uint4*)dst = *(const uint4*)src;
#endif
}

template <int N>
static __device__ inline void wait_async() {
#if ASYNC_STAGE
#if __has_builtin(__builtin_amdgcn_s_wait_asynccnt)
    __builtin_amdgcn_s_wait_asynccnt((unsigned short)N);
#else
    asm volatile("s_wait_asynccnt %0" ::"n"(N));
#endif
#endif
}

// ---------------------------------------------------------------------------
// WMMA fragment loaders (16-bit A 16x32 and B 32x16 layouts per CDNA5 ISA)
//   A: lane l -> row M = l&15 ; half-wave selects K interleave {0..7,16..23}
//      vs {8..15,24..31}
//   B: lane l -> col N = l&15 ; half-wave selects K interleave {0..7,16..23}
//      vs {8..15,24..31}
// Both reduce to two contiguous 16B LDS loads per lane when the tile is
// stored with K contiguous (A row-major tile / B^T tile).
// ---------------------------------------------------------------------------
static __device__ inline half16 frag_combine(half8 lo, half8 hi) {
    return __builtin_shufflevector(lo, hi, 0, 1, 2, 3, 4, 5, 6, 7,
                                           8, 9, 10, 11, 12, 13, 14, 15);
}

static __device__ inline half16 ld_fragA(const _Float16* base, int stride) {
    int l  = threadIdx.x & 31;
    int r  = l & 15;
    int kb = (l < 16) ? 0 : 8;
    const _Float16* p = base + r * stride + kb;
    half8 lo = *(const half8*)(p);
    half8 hi = *(const half8*)(p + 16);
    return frag_combine(lo, hi);
}

static __device__ inline half16 ld_fragB(const _Float16* base, int stride) {
    int l  = threadIdx.x & 31;
    int r  = l & 15;
    int kb = (l < 16) ? 0 : 8;
    const _Float16* p = base + r * stride + kb;
    half8 lo = *(const half8*)(p);
    half8 hi = *(const half8*)(p + 16);
    return frag_combine(lo, hi);
}

static __device__ inline floatx8 wmma_f16(half16 a, half16 b, floatx8 c) {
    return __builtin_amdgcn_wmma_f32_16x16x32_f16(false, a, false, b,
                                                  (short)0, c, false, false);
}

// ---------------------------------------------------------------------------
// Generic tiled WMMA GEMM:  C[M,N] = A[M,K](f16, row-major) x Bt[N,K](f16)
// 128x128 C-tile per workgroup, 8 waves (2x4), wave tile 64x32.
// BK = 64, double-buffered LDS, async global->LDS staging one tile ahead.
// MODE: 0 = f32 out (+bias)
//       1 = f16 out (+bias)
//       2 = f32+f16 out with residual add (+bias)
//       3 = exact GELU -> f16 out (+bias)
//       4 = f32 out += comb[row]*(acc+bias)      (MoE expert combine)
// ---------------------------------------------------------------------------
template <int MODE>
__global__ __launch_bounds__(256) void gemm_wmma(
    const _Float16* __restrict__ A, const _Float16* __restrict__ Bt,
    const float* __restrict__ bias, const float* __restrict__ res,
    const float* __restrict__ comb, int combStride,
    float* __restrict__ outF32, _Float16* __restrict__ outF16,
    int M, int N, int K)
{
    __shared__ __align__(16) _Float16 As[2][128][72];  // +8 pad: conflict-free
    __shared__ __align__(16) _Float16 Bs[2][128][72];

    const int row0 = blockIdx.x * 128;
    const int col0 = blockIdx.y * 128;
    const int tid  = threadIdx.x;
    const int wv   = tid >> 5;
    const int wm   = wv & 1;        // 2 wave-rows of 64
    const int wn   = wv >> 1;       // 4 wave-cols of 32

    floatx8 acc[4][2] = {};

    const int r    = tid >> 1;      // 0..127 staged row
    const int part = tid & 1;       // 32-half segment
    const _Float16* gA = A  + (size_t)(row0 + r) * K + part * 32;
    const _Float16* gB = Bt + (size_t)(col0 + r) * K + part * 32;

    // stage one 128x64 tile of A and of Bt into buffer `buf` (8 async b128/thread)
    auto stage = [&](int buf, int kt) {
        _Float16* dA = &As[buf][r][part * 32];
        _Float16* dB = &Bs[buf][r][part * 32];
#pragma unroll
        for (int c = 0; c < 4; ++c) {
            stage16(dA + c * 8, gA + kt + c * 8);
            stage16(dB + c * 8, gB + kt + c * 8);
        }
    };

    const int ntiles = K >> 6;
    stage(0, 0);
    for (int t = 0; t < ntiles; ++t) {
        const int cur = t & 1;
        if (t + 1 < ntiles) {
            stage(cur ^ 1, (t + 1) * 64);   // run-ahead copy into other buffer
            wait_async<8>();                // retire current tile's 8 async ops
        } else {
            wait_async<0>();
        }
        __syncthreads();
#pragma unroll
        for (int kh = 0; kh < 2; ++kh) {
#pragma unroll
            for (int mt = 0; mt < 4; ++mt) {
                half16 a = ld_fragA(&As[cur][wm * 64 + mt * 16][kh * 32], 72);
#pragma unroll
                for (int nt = 0; nt < 2; ++nt) {
                    half16 b = ld_fragB(&Bs[cur][wn * 32 + nt * 16][kh * 32], 72);
                    acc[mt][nt] = wmma_f16(a, b, acc[mt][nt]);
                }
            }
        }
        __syncthreads();
    }

    // C layout: lane l -> N = l&15, M-base = (l>=16)*8, vgpr r -> M row
    const int l  = tid & 31;
    const int n  = l & 15;
    const int mb = (l & 16) ? 8 : 0;
#pragma unroll
    for (int mt = 0; mt < 4; ++mt) {
#pragma unroll
        for (int nt = 0; nt < 2; ++nt) {
#pragma unroll
            for (int rr = 0; rr < 8; ++rr) {
                int row = row0 + wm * 64 + mt * 16 + mb + rr;
                int col = col0 + wn * 32 + nt * 16 + n;
                size_t idx = (size_t)row * N + col;
                float v = acc[mt][nt][rr];
                if (bias) v += bias[col];
                if constexpr (MODE == 0) {
                    outF32[idx] = v;
                } else if constexpr (MODE == 1) {
                    outF16[idx] = (_Float16)v;
                } else if constexpr (MODE == 2) {
                    float t2 = v + res[idx];
                    outF32[idx] = t2;
                    outF16[idx] = (_Float16)t2;
                } else if constexpr (MODE == 3) {
                    float g = 0.5f * v * (1.0f + erff(v * 0.70710678118f));
                    outF16[idx] = (_Float16)g;
                } else {
                    outF32[idx] += comb[(size_t)row * combStride] * v;
                }
            }
        }
    }
}

// ---------------------------------------------------------------------------
// Wave-modulated attention, one workgroup per (batch, head).
// scores = (q @ k^T) * DK^-0.5 * cos(2*pi*f*pos + phi) ; softmax ; @ v
// Both matmuls run on WMMA; softmax in f32 via wave32 half-shuffles.
// ---------------------------------------------------------------------------
__global__ __launch_bounds__(256) void attn_wave(
    const _Float16* __restrict__ q, const _Float16* __restrict__ k,
    const _Float16* __restrict__ v, _Float16* __restrict__ ao,
    const float* __restrict__ freq, const float* __restrict__ phase)
{
    __shared__ __align__(16) _Float16 Qs[128][40];
    __shared__ __align__(16) _Float16 Ks[128][40];
    __shared__ __align__(16) _Float16 Vt[32][136];
    __shared__ __align__(16) _Float16 Ps[128][136];

    const int hh  = blockIdx.x & (HH - 1);
    const int b   = blockIdx.x >> 3;
    const int tid = threadIdx.x;
    const int w   = tid >> 5;
    const float f  = freq[hh];
    const float ph = phase[hh];

    // stage Q/K rows (each: 128 rows x 32 halfs)
    {
        const int r = tid >> 1, part = tid & 1;
        size_t g = (size_t)(b * LL + r) * DD + hh * DK + part * 16;
        *(uint4*)&Qs[r][part * 16]     = *(const uint4*)(q + g);
        *(uint4*)&Qs[r][part * 16 + 8] = *(const uint4*)(q + g + 8);
        *(uint4*)&Ks[r][part * 16]     = *(const uint4*)(k + g);
        *(uint4*)&Ks[r][part * 16 + 8] = *(const uint4*)(k + g + 8);
    }
    // stage V transposed: Vt[dk][l]
    for (int idx = tid; idx < 128 * 32; idx += 256) {
        int lrow = idx >> 5, dk = idx & 31;
        Vt[dk][lrow] = v[(size_t)(b * LL + lrow) * DD + hh * DK + dk];
    }
    __syncthreads();

    // ---- scores: wave w owns query rows [w*16, w*16+16), all 128 keys ----
    floatx8 sc[8] = {};
    {
        half16 aq = ld_fragA(&Qs[w * 16][0], 40);
#pragma unroll
        for (int t = 0; t < 8; ++t) {
            half16 bk = ld_fragB(&Ks[t * 16][0], 40);
            sc[t] = wmma_f16(aq, bk, sc[t]);
        }
    }

    const int l  = tid & 31;
    const int n  = l & 15;
    const int mb = (l & 16) ? 8 : 0;

    // fold in DK^-0.5 and the per-key cosine wave
#pragma unroll
    for (int t = 0; t < 8; ++t) {
        float pos = (float)(t * 16 + n);
        float wf  = 0.17677669529f * __cosf(6.28318530718f * f * pos + ph);
#pragma unroll
        for (int rr = 0; rr < 8; ++rr) sc[t][rr] *= wf;
    }

    // softmax per row (row value lives in lanes {l&15} of one 16-lane half)
#pragma unroll
    for (int rr = 0; rr < 8; ++rr) {
        float m = sc[0][rr];
#pragma unroll
        for (int t = 1; t < 8; ++t) m = fmaxf(m, sc[t][rr]);
        for (int off = 8; off >= 1; off >>= 1)
            m = fmaxf(m, __shfl_xor(m, off, 16));
        float s = 0.f;
#pragma unroll
        for (int t = 0; t < 8; ++t) {
            float e = __expf(sc[t][rr] - m);
            sc[t][rr] = e;
            s += e;
        }
        for (int off = 8; off >= 1; off >>= 1)
            s += __shfl_xor(s, off, 16);
        float inv = 1.0f / s;
        int row = w * 16 + mb + rr;
#pragma unroll
        for (int t = 0; t < 8; ++t)
            Ps[row][t * 16 + n] = (_Float16)(sc[t][rr] * inv);
    }
    __syncthreads();

    // ---- out = probs @ v : M=16 rows per wave, N=32, K=128 (4 steps) ----
    floatx8 o[2] = {};
#pragma unroll
    for (int kc = 0; kc < 4; ++kc) {
        half16 ap = ld_fragA(&Ps[w * 16][kc * 32], 136);
#pragma unroll
        for (int nt = 0; nt < 2; ++nt) {
            half16 bv = ld_fragB(&Vt[nt * 16][kc * 32], 136);
            o[nt] = wmma_f16(ap, bv, o[nt]);
        }
    }
#pragma unroll
    for (int nt = 0; nt < 2; ++nt)
#pragma unroll
        for (int rr = 0; rr < 8; ++rr) {
            int row = w * 16 + mb + rr;
            int col = nt * 16 + n;
            ao[(size_t)(b * LL + row) * DD + hh * DK + col] = (_Float16)o[nt][rr];
        }
}

// ---------------------------------------------------------------------------
// Small helper kernels
// ---------------------------------------------------------------------------
__global__ void embed_kernel(const float* __restrict__ x,
                             const float* __restrict__ ew,
                             const float* __restrict__ eb,
                             float* __restrict__ hf, _Float16* __restrict__ hh_)
{
    int idx = blockIdx.x * blockDim.x + threadIdx.x;
    if (idx >= TT * DD) return;
    int d = idx & (DD - 1);
    int t = idx >> 8;
    float a = eb[d];
#pragma unroll
    for (int i = 0; i < DIN; ++i) a += x[t * DIN + i] * ew[i * DD + d];
    hf[idx]  = a;
    hh_[idx] = (_Float16)a;
}

__global__ void cvt_f16(const float* __restrict__ src,
                        _Float16* __restrict__ dst, size_t ncount)
{
    for (size_t i = blockIdx.x * (size_t)blockDim.x + threadIdx.x; i < ncount;
         i += (size_t)gridDim.x * blockDim.x)
        dst[i] = (_Float16)src[i];
}

// f32 [batch][K][N] -> f16 [batch][N][K]   (B^T layout for WMMA B fragments)
__global__ void cvt_f16_T(const float* __restrict__ src,
                          _Float16* __restrict__ dst, int batch, int K, int N)
{
    size_t total = (size_t)batch * K * N;
    for (size_t i = blockIdx.x * (size_t)blockDim.x + threadIdx.x; i < total;
         i += (size_t)gridDim.x * blockDim.x) {
        int n = (int)(i % N);
        size_t bk = i / N;
        int kk = (int)(bk % K);
        int bb = (int)(bk / K);
        dst[((size_t)bb * N + n) * K + kk] = (_Float16)src[i];
    }
}

// Router: softmax over NEXP logits, top-2, renormalize -> dense comb[T,NEXP]
__global__ void router_kernel(const float* __restrict__ h,
                              const float* __restrict__ rw,
                              const float* __restrict__ rb,
                              float* __restrict__ comb)
{
    int t = blockIdx.x * blockDim.x + threadIdx.x;
    if (t >= TT) return;
    const float* x = h + (size_t)t * DD;
    float lg[NEXP];
#pragma unroll
    for (int e = 0; e < NEXP; ++e) lg[e] = rb[e];
    for (int d = 0; d < DD; ++d) {
        float xv = x[d];
        const float* wrow = rw + d * NEXP;
#pragma unroll
        for (int e = 0; e < NEXP; ++e) lg[e] += xv * wrow[e];
    }
    float m = lg[0];
#pragma unroll
    for (int e = 1; e < NEXP; ++e) m = fmaxf(m, lg[e]);
    float s = 0.f;
#pragma unroll
    for (int e = 0; e < NEXP; ++e) { lg[e] = __expf(lg[e] - m); s += lg[e]; }
    float inv = 1.0f / s;
#pragma unroll
    for (int e = 0; e < NEXP; ++e) lg[e] *= inv;
    int i0 = 0;
#pragma unroll
    for (int e = 1; e < NEXP; ++e) if (lg[e] > lg[i0]) i0 = e;
    int i1 = (i0 == 0) ? 1 : 0;
#pragma unroll
    for (int e = 0; e < NEXP; ++e)
        if (e != i0 && lg[e] > lg[i1]) i1 = e;
    float wsum = 1.0f / (lg[i0] + lg[i1]);
    float* c = comb + (size_t)t * NEXP;
#pragma unroll
    for (int e = 0; e < NEXP; ++e) c[e] = 0.f;
    c[i0] = lg[i0] * wsum;
    c[i1] = lg[i1] * wsum;
}

// Final LayerNorm (last token only) + prediction / softplus-uncertainty heads
__global__ __launch_bounds__(256) void head_kernel(
    const float* __restrict__ h, const float* __restrict__ ln_g,
    const float* __restrict__ ln_b, const float* __restrict__ pw,
    const float* __restrict__ pb, const float* __restrict__ uw,
    const float* __restrict__ ub, float* __restrict__ out)
{
    __shared__ float red[256];
    __shared__ float lat[256];
    int b = blockIdx.x, tid = threadIdx.x;
    float v = h[((size_t)b * LL + (LL - 1)) * DD + tid];

    red[tid] = v;
    __syncthreads();
    for (int s = 128; s > 0; s >>= 1) {
        if (tid < s) red[tid] += red[tid + s];
        __syncthreads();
    }
    float mu = red[0] * (1.0f / DD);
    __syncthreads();
    float d = v - mu;
    red[tid] = d * d;
    __syncthreads();
    for (int s = 128; s > 0; s >>= 1) {
        if (tid < s) red[tid] += red[tid + s];
        __syncthreads();
    }
    float var = red[0] * (1.0f / DD);
    __syncthreads();
    lat[tid] = d * rsqrtf(var + 1e-5f) * ln_g[tid] + ln_b[tid];
    __syncthreads();

    if (tid < 2 * NOUTP) {
        int n = tid % NOUTP;
        int u = tid / NOUTP;
        const float* W = u ? uw : pw;
        float a = u ? ub[n] : pb[n];
        for (int dd = 0; dd < DD; ++dd) a += lat[dd] * W[dd * NOUTP + n];
        if (u) a = (a > 20.f) ? a : log1pf(__expf(a));
        out[u * (BB * NOUTP) + b * NOUTP + n] = a;
    }
}

// ---------------------------------------------------------------------------
// Host orchestration
// ---------------------------------------------------------------------------
static inline uint8_t* carve(uint8_t*& p, size_t bytes) {
    uint8_t* r = p;
    p += (bytes + 255) & ~(size_t)255;
    return r;
}

extern "C" void kernel_launch(void* const* d_in, const int* in_sizes, int n_in,
                              void* d_out, int out_size, void* d_ws, size_t ws_size,
                              hipStream_t stream) {
    (void)in_sizes; (void)n_in; (void)out_size; (void)ws_size;
    const float* x      = (const float*)d_in[0];
    const float* emb_w  = (const float*)d_in[1];
    const float* emb_b  = (const float*)d_in[2];
    const float* wq     = (const float*)d_in[3];
    const float* wk     = (const float*)d_in[4];
    const float* wv     = (const float*)d_in[5];
    const float* wo     = (const float*)d_in[6];
    const float* wfreq  = (const float*)d_in[7];
    const float* wphase = (const float*)d_in[8];
    const float* rtw    = (const float*)d_in[9];
    const float* rtb    = (const float*)d_in[10];
    const float* ew1    = (const float*)d_in[11];
    const float* eb1    = (const float*)d_in[12];
    const float* ew2    = (const float*)d_in[13];
    const float* eb2    = (const float*)d_in[14];
    const float* ln_g   = (const float*)d_in[15];
    const float* ln_b   = (const float*)d_in[16];
    const float* pred_w = (const float*)d_in[17];
    const float* pred_b = (const float*)d_in[18];
    const float* unc_w  = (const float*)d_in[19];
    const float* unc_b  = (const float*)d_in[20];

    uint8_t* p = (uint8_t*)d_ws;
    float*     hF32  = (float*)    carve(p, (size_t)TT * DD * 4);
    _Float16*  hF16  = (_Float16*) carve(p, (size_t)TT * DD * 2);
    _Float16*  qF16  = (_Float16*) carve(p, (size_t)TT * DD * 2);
    _Float16*  kF16  = (_Float16*) carve(p, (size_t)TT * DD * 2);
    _Float16*  vF16  = (_Float16*) carve(p, (size_t)TT * DD * 2);
    _Float16*  aoF16 = (_Float16*) carve(p, (size_t)TT * DD * 2);
    _Float16*  h1F16 = (_Float16*) carve(p, (size_t)TT * DFF * 2);
    float*     comb  = (float*)    carve(p, (size_t)TT * NEXP * 4);
    _Float16*  wqT   = (_Float16*) carve(p, (size_t)NLAY * DD * DD * 2);
    _Float16*  wkT   = (_Float16*) carve(p, (size_t)NLAY * DD * DD * 2);
    _Float16*  wvT   = (_Float16*) carve(p, (size_t)NLAY * DD * DD * 2);
    _Float16*  woT   = (_Float16*) carve(p, (size_t)NLAY * DD * DD * 2);
    _Float16*  e1T   = (_Float16*) carve(p, (size_t)NMOE * NEXP * DD * DFF * 2);
    _Float16*  e2T   = (_Float16*) carve(p, (size_t)NMOE * NEXP * DFF * DD * 2);

    // Convert + pre-transpose all weights to f16 B^T layout
    cvt_f16_T<<<2048, 256, 0, stream>>>(wq,  wqT, NLAY, DD, DD);
    cvt_f16_T<<<2048, 256, 0, stream>>>(wk,  wkT, NLAY, DD, DD);
    cvt_f16_T<<<2048, 256, 0, stream>>>(wv,  wvT, NLAY, DD, DD);
    cvt_f16_T<<<2048, 256, 0, stream>>>(wo,  woT, NLAY, DD, DD);
    cvt_f16_T<<<4096, 256, 0, stream>>>(ew1, e1T, NMOE * NEXP, DD, DFF);
    cvt_f16_T<<<4096, 256, 0, stream>>>(ew2, e2T, NMOE * NEXP, DFF, DD);

    embed_kernel<<<(TT * DD + 255) / 256, 256, 0, stream>>>(x, emb_w, emb_b,
                                                            hF32, hF16);

    const dim3 gD(TT / 128, DD / 128);   // N=256 GEMMs
    const dim3 gF(TT / 128, DFF / 128);  // N=1024 GEMMs

    for (int i = 0; i < NLAY; ++i) {
        const size_t wOff = (size_t)i * DD * DD;
        gemm_wmma<1><<<gD, 256, 0, stream>>>(hF16, wqT + wOff, nullptr, nullptr,
                                             nullptr, 0, nullptr, qF16, TT, DD, DD);
        gemm_wmma<1><<<gD, 256, 0, stream>>>(hF16, wkT + wOff, nullptr, nullptr,
                                             nullptr, 0, nullptr, kF16, TT, DD, DD);
        gemm_wmma<1><<<gD, 256, 0, stream>>>(hF16, wvT + wOff, nullptr, nullptr,
                                             nullptr, 0, nullptr, vF16, TT, DD, DD);
        attn_wave<<<BB * HH, 256, 0, stream>>>(qF16, kF16, vF16, aoF16,
                                               wfreq + i * HH, wphase + i * HH);
        // output projection + residual (f32 chain), refresh f16 mirror
        gemm_wmma<2><<<gD, 256, 0, stream>>>(aoF16, woT + wOff, nullptr, hF32,
                                             nullptr, 0, hF32, hF16, TT, DD, DD);

        if ((i % 2) == 0 && (i / 2) < NMOE) {
            int m = i / 2;
            router_kernel<<<TT / 256, 256, 0, stream>>>(
                hF32, rtw + (size_t)m * DD * NEXP, rtb + m * NEXP, comb);
            for (int e = 0; e < NEXP; ++e) {
                const size_t w1Off = ((size_t)m * NEXP + e) * DD * DFF;
                const size_t w2Off = ((size_t)m * NEXP + e) * DFF * DD;
                gemm_wmma<3><<<gF, 256, 0, stream>>>(
                    hF16, e1T + w1Off, eb1 + ((size_t)m * NEXP + e) * DFF,
                    nullptr, nullptr, 0, nullptr, h1F16, TT, DFF, DD);
                gemm_wmma<4><<<gD, 256, 0, stream>>>(
                    h1F16, e2T + w2Off, eb2 + ((size_t)m * NEXP + e) * DD,
                    nullptr, comb + e, NEXP, hF32, nullptr, TT, DD, DFF);
            }
            cvt_f16<<<2048, 256, 0, stream>>>(hF32, hF16, (size_t)TT * DD);
        }
    }

    head_kernel<<<BB, 256, 0, stream>>>(hF32, ln_g, ln_b, pred_w, pred_b,
                                        unc_w, unc_b, (float*)d_out);
}
